// MultiTaskGAT_13984413515973
// MI455X (gfx1250) — matrix operand order repaired and checked by reading, not compiled, not run on hardware
//
#include <hip/hip_runtime.h>
#include <hip/hip_bf16.h>

#define BATCH   4096
#define NSQ     64
#define CIN     17
#define HDIM    128
#define NHEADS  8
#define GDEPTH  4
#define HEADDIM 16
#define LOG2E   1.44269504088896f

#if __has_builtin(__builtin_amdgcn_exp2f)
#define EXP2F(x) __builtin_amdgcn_exp2f(x)
#else
#define EXP2F(x) exp2f(x)
#endif

typedef __attribute__((ext_vector_type(16))) __bf16       v16bf;
typedef __attribute__((ext_vector_type(8)))  float        v8f;
typedef __attribute__((ext_vector_type(4)))  unsigned int u32x4;

union ATile { u32x4 u[2]; v16bf v; };
union BPack { u32x4 u; __bf16 h[8]; };

__global__ __launch_bounds__(256) void gat_fused_kernel(
    const float* __restrict__ x,     const float* __restrict__ adj,
    const float* __restrict__ W_in,  const float* __restrict__ b_in,
    const float* __restrict__ Wg,    const float* __restrict__ a_src,
    const float* __restrict__ a_dst, const float* __restrict__ W_m,
    const float* __restrict__ b_m,   const float* __restrict__ W_c1,
    const float* __restrict__ b_c1,  const float* __restrict__ W_c2,
    const float* __restrict__ b_c2,
    float* __restrict__ out_emb, float* __restrict__ out_logit)
{
  // ---- LDS (~141 KB; two workgroups fit in the 320 KB WGP) ----
  __shared__ __align__(16) float  sh[NSQ][HDIM];        // 32 KB f32 features (residual/pool)
  __shared__ __align__(16) __bf16 sh_bf[NSQ][HDIM];     // 16 KB bf16 mirror  (WMMA A, GEMM)
  __shared__ __align__(16) __bf16 shpT_bf[HDIM][NSQ];   // 16 KB h'^T = (h@Wg)^T (WMMA B, agg)
  __shared__ __align__(16) __bf16 sattn[NHEADS][NSQ][NSQ]; // 64 KB unnormalized exp scores
  __shared__ float sed[NHEADS][NSQ];                    // log2e-scaled dst logits
  __shared__ float srcp[NHEADS][NSQ];                   // 1/rowsum
  __shared__ unsigned long long sadj[NSQ];
  __shared__ float sx[CIN][NSQ];
  __shared__ float sg[HDIM];
  __shared__ float semb[HDIM];
  __shared__ float sq[NSQ];
  __shared__ float snorm;

  const int tid  = threadIdx.x;
  const int wave = tid >> 5;      // 8 waves of 32 (wave32)
  const int lane = tid & 31;
  const int l15  = lane & 15;
  const int half = lane >> 4;
  const int b    = blockIdx.x;

  // ---- stage x[b] and adjacency bitmask ----
  for (int idx = tid; idx < CIN * NSQ; idx += 256)
    ((float*)sx)[idx] = x[(size_t)b * CIN * NSQ + idx];
  if (tid < NSQ) {
    unsigned long long m = 0ull;
    for (int j = 0; j < NSQ; ++j)
      if (adj[tid * NSQ + j] > 0.f) m |= (1ull << j);
    sadj[tid] = m;
  }
  __syncthreads();

  // ---- input projection: h[i][c] = sum_ci x[b][ci][i] * W_in[ci][c] + b_in[c] ----
  for (int r = 0; r < (NSQ * HDIM) / 256; ++r) {
    int flat = r * 256 + tid;
    int i = flat >> 7, c = flat & 127;
    float acc = b_in[c];
#pragma unroll
    for (int ci = 0; ci < CIN; ++ci)
      acc = fmaf(sx[ci][i], W_in[ci * HDIM + c], acc);
    sh[i][c]    = acc;
    sh_bf[i][c] = (__bf16)acc;
  }
  __syncthreads();

  // ============================ GAT layers ============================
  for (int l = 0; l < GDEPTH; ++l) {
    const float* W   = Wg + (size_t)l * HDIM * HDIM;
    const int   head = wave;
    const int   n0   = wave * 16;

    // ---- GEMM: hp = h @ W   (M=64, N=128, K=128; wave owns one N-tile) ----
    v8f acc0 = {}, acc1 = {}, acc2 = {}, acc3 = {};
#pragma unroll
    for (int kt = 0; kt < 4; ++kt) {
      const int k0 = kt * 32;
      // B tile (32x16) from L2-resident Wg: lane = column n0+l15; halves split K
      v16bf bm;
#pragma unroll
      for (int v = 0; v < 8; ++v) {
        int kb = k0 + ((v < 4) ? (half * 8 + 2 * v) : (16 + half * 8 + 2 * (v - 4)));
        bm[2 * v]     = (__bf16)W[kb * HDIM + n0 + l15];
        bm[2 * v + 1] = (__bf16)W[(kb + 1) * HDIM + n0 + l15];
      }
#pragma unroll
      for (int mt = 0; mt < 4; ++mt) {
        const int m = mt * 16 + l15;   // A: lane = row, two contiguous 8-elem K chunks
        ATile a;
        a.u[0] = *(const u32x4*)&sh_bf[m][k0 + half * 8];
        a.u[1] = *(const u32x4*)&sh_bf[m][k0 + 16 + half * 8];
        v8f& acc = (mt == 0) ? acc0 : (mt == 1) ? acc1 : (mt == 2) ? acc2 : acc3;
        acc = __builtin_amdgcn_wmma_f32_16x16x32_bf16(false, a.v, false, bm,
                                                      (short)0, acc, false, false);
      }
    }
    // prefetch next layer's weights while we do attention math
    if (l + 1 < GDEPTH)
      __builtin_prefetch(Wg + (size_t)(l + 1) * HDIM * HDIM + tid * 64, 0, 1);
    // D layout: lane = column n0+l15; VGPR v -> row half*8+v.
    // Rows are contiguous in hpT -> one packed b128 store per M-tile.
#pragma unroll
    for (int mt = 0; mt < 4; ++mt) {
      const v8f& acc = (mt == 0) ? acc0 : (mt == 1) ? acc1 : (mt == 2) ? acc2 : acc3;
      BPack p;
#pragma unroll
      for (int v = 0; v < 8; ++v) p.h[v] = (__bf16)acc[v];
      *(u32x4*)&shpT_bf[n0 + l15][mt * 16 + half * 8] = p.u;
    }
    __syncthreads();

    // ---- per-head attention logits (log2e-scaled); es stays in registers ----
    float asv[HEADDIM], adv[HEADDIM];
#pragma unroll
    for (int k = 0; k < HEADDIM; ++k) {
      asv[k] = a_src[((size_t)l * NHEADS + head) * HEADDIM + k];
      adv[k] = a_dst[((size_t)l * NHEADS + head) * HEADDIM + k];
    }
    float esr[2];
#pragma unroll
    for (int p = 0; p < 2; ++p) {
      int i = lane + 32 * p;
      float es = 0.f, ed = 0.f;
#pragma unroll
      for (int k = 0; k < HEADDIM; ++k) {
        float hv = (float)shpT_bf[head * HEADDIM + k][i];
        es = fmaf(hv, asv[k], es);
        ed = fmaf(hv, adv[k], ed);
      }
      esr[p] = es * LOG2E;          // producer lane == consumer lane
      sed[head][i] = ed * LOG2E;    // read by all lanes of this wave
    }
    __syncthreads();

    // maxall(ed) per head via wave reduction: valid shift bound for softmax
    float medv = fmaxf(sed[head][lane], sed[head][lane + 32]);
#pragma unroll
    for (int off = 16; off > 0; off >>= 1)
      medv = fmaxf(medv, __shfl_xor(medv, off, 32));

    // ---- single-pass masked leaky-ReLU softmax (base-2 domain) ----
    for (int p = 0; p < 2; ++p) {
      int i = lane + 32 * p;
      float esi = esr[p];
      unsigned long long msk = sadj[i];
      float ub = esi + medv;
      float C  = fmaxf(ub, 0.2f * ub);  // leaky(upper bound) >= all row scores
      float s  = 0.f;
      for (int jc = 0; jc < NSQ; jc += 8) {
        BPack pk;
#pragma unroll
        for (int jj = 0; jj < 8; ++jj) {
          int j = jc + jj;
          float e  = esi + sed[head][j];
          float lk = fmaxf(e, 0.2f * e);        // leaky_relu
          float w  = EXP2F(lk - C);             // <= 1, no overflow
          w = ((msk >> j) & 1ull) ? w : 0.f;
          s += w;
          pk.h[jj] = (__bf16)w;
        }
        *(u32x4*)&sattn[head][i][jc] = pk.u;
      }
      srcp[head][i] = __builtin_amdgcn_rcpf(s);
    }
    __syncthreads();

    // ---- aggregation: out_head = attn @ hp_head  (M=64, N=16, K=64) ----
    v8f g0 = {}, g1 = {}, g2 = {}, g3 = {};
#pragma unroll
    for (int kt = 0; kt < 2; ++kt) {
      const int k0 = kt * 32;
      // B tile: lane = channel head*16+l15; two contiguous 8-elem K chunks in hpT -> 2x b128
      ATile bt;
      bt.u[0] = *(const u32x4*)&shpT_bf[head * HEADDIM + l15][k0 + half * 8];
      bt.u[1] = *(const u32x4*)&shpT_bf[head * HEADDIM + l15][k0 + 16 + half * 8];
#pragma unroll
      for (int mt = 0; mt < 4; ++mt) {
        const int m = mt * 16 + l15;
        ATile a;
        a.u[0] = *(const u32x4*)&sattn[head][m][k0 + half * 8];
        a.u[1] = *(const u32x4*)&sattn[head][m][k0 + 16 + half * 8];
        v8f& acc = (mt == 0) ? g0 : (mt == 1) ? g1 : (mt == 2) ? g2 : g3;
        acc = __builtin_amdgcn_wmma_f32_16x16x32_bf16(false, a.v, false, bt.v,
                                                      (short)0, acc, false, false);
      }
    }
    // epilogue: normalize by 1/rowsum, residual, ELU, refresh mirrors
#pragma unroll
    for (int mt = 0; mt < 4; ++mt) {
      const v8f& acc = (mt == 0) ? g0 : (mt == 1) ? g1 : (mt == 2) ? g2 : g3;
#pragma unroll
      for (int v = 0; v < 8; ++v) {
        int row = mt * 16 + half * 8 + v;
        int cc  = head * HEADDIM + l15;
        float o  = acc[v] * srcp[head][row];
        float nh = sh[row][cc] + o;
        nh = (nh > 0.f) ? nh : (EXP2F(nh * LOG2E) - 1.f);   // ELU
        sh[row][cc]    = nh;
        sh_bf[row][cc] = (__bf16)nh;
      }
    }
    __syncthreads();
  }

  // ---- global max pool over nodes ----
  if (tid < HDIM) {
    float g = -3.0e38f;
    for (int i = 0; i < NSQ; ++i) g = fmaxf(g, sh[i][tid]);
    sg[tid] = g;
  }
  __syncthreads();

  // ---- heads (split across thread ranges) ----
  if (tid < HDIM) {                       // metric embedding
    float e = b_m[tid];
    for (int k = 0; k < HDIM; ++k) e = fmaf(sg[k], W_m[k * HDIM + tid], e);
    semb[tid] = e;
  } else if (tid < HDIM + 64) {           // value-head hidden + ReLU
    int t = tid - HDIM;
    float q = b_c1[t];
    for (int k = 0; k < HDIM; ++k) q = fmaf(sg[k], W_c1[k * 64 + t], q);
    sq[t] = fmaxf(q, 0.f);
  }
  __syncthreads();
  if (tid == 0) {
    float ss = 0.f;
    for (int c = 0; c < HDIM; ++c) ss = fmaf(semb[c], semb[c], ss);
    snorm = __builtin_amdgcn_rcpf(fmaxf(sqrtf(ss), 1e-12f));
    float lg = b_c2[0];
    for (int t = 0; t < 64; ++t) lg = fmaf(sq[t], W_c2[t], lg);
    out_logit[b] = lg;
  }
  __syncthreads();
  if (tid < HDIM)
    out_emb[(size_t)b * HDIM + tid] = semb[tid] * snorm;
}

extern "C" void kernel_launch(void* const* d_in, const int* in_sizes, int n_in,
                              void* d_out, int out_size, void* d_ws, size_t ws_size,
                              hipStream_t stream) {
  (void)in_sizes; (void)n_in; (void)out_size; (void)d_ws; (void)ws_size;
  const float* x     = (const float*)d_in[0];
  const float* adj   = (const float*)d_in[1];
  const float* W_in  = (const float*)d_in[2];
  const float* b_in  = (const float*)d_in[3];
  const float* Wg    = (const float*)d_in[4];
  const float* a_src = (const float*)d_in[5];
  const float* a_dst = (const float*)d_in[6];
  const float* W_m   = (const float*)d_in[7];
  const float* b_m   = (const float*)d_in[8];
  const float* W_c1  = (const float*)d_in[9];
  const float* b_c1  = (const float*)d_in[10];
  const float* W_c2  = (const float*)d_in[11];
  const float* b_c2  = (const float*)d_in[12];

  float* out       = (float*)d_out;
  float* out_emb   = out;                          // [4096,128]
  float* out_logit = out + (size_t)BATCH * HDIM;   // [4096,1]

  gat_fused_kernel<<<BATCH, 256, 0, stream>>>(
      x, adj, W_in, b_in, Wg, a_src, a_dst, W_m, b_m, W_c1, b_c1, W_c2, b_c2,
      out_emb, out_logit);
}
